// Block_13005160972488
// MI455X (gfx1250) — matrix operand order repaired and checked by reading, not compiled, not run on hardware
//
#include <hip/hip_runtime.h>
#include <hip/hip_bf16.h>

typedef __attribute__((ext_vector_type(16))) __bf16 v16bf;
typedef __attribute__((ext_vector_type(8)))  __bf16 v8bf;
typedef __attribute__((ext_vector_type(8)))  float  v8f;

#define D_MODEL 768
#define SEQ 512
#define NH 12
#define DH 64

// ---------- helpers ----------
static __device__ inline v16bf ldfrag(const __bf16* lo, const __bf16* hi) {
    v8bf a = *(const v8bf*)lo;
    v8bf b = *(const v8bf*)hi;
    v16bf r;
#pragma unroll
    for (int i = 0; i < 8; ++i) { r[i] = a[i]; r[i + 8] = b[i]; }
    return r;
}

// 8 consecutive f32 (16B-aligned) -> packed bf16 -> single 16B LDS store
static __device__ inline void stage8(const float* __restrict__ g, __bf16* dst) {
    float4 f0 = *(const float4*)g;
    float4 f1 = *(const float4*)(g + 4);
    v8bf r;
    r[0] = (__bf16)f0.x; r[1] = (__bf16)f0.y; r[2] = (__bf16)f0.z; r[3] = (__bf16)f0.w;
    r[4] = (__bf16)f1.x; r[5] = (__bf16)f1.y; r[6] = (__bf16)f1.z; r[7] = (__bf16)f1.w;
    *(v8bf*)dst = r;
}

static __device__ inline float redmax16(float v) {
#pragma unroll
    for (int m = 1; m < 16; m <<= 1) v = fmaxf(v, __shfl_xor(v, m, 32));
    return v;
}
static __device__ inline float redsum16(float v) {
#pragma unroll
    for (int m = 1; m < 16; m <<= 1) v += __shfl_xor(v, m, 32);
    return v;
}

// ---------- gaussian positional bias: gbias[i][j] = softmax_j(-0.5*(j-i)^2) ----------
__global__ __launch_bounds__(256) void gbias_kernel(float* __restrict__ gbias) {
    __shared__ float ebuf[SEQ];
    __shared__ float red[256];
    int i = blockIdx.x, tid = threadIdx.x;
    float acc = 0.f;
#pragma unroll
    for (int r = 0; r < 2; ++r) {
        int j = tid + r * 256;
        float d = (float)(j - i);
        float e = __expf(-0.5f * d * d);
        ebuf[j] = e;
        acc += e;
    }
    red[tid] = acc;
    __syncthreads();
    for (int s = 128; s > 0; s >>= 1) {
        if (tid < s) red[tid] += red[tid + s];
        __syncthreads();
    }
    float inv = 1.f / red[0];
#pragma unroll
    for (int r = 0; r < 2; ++r) {
        int j = tid + r * 256;
        gbias[(size_t)i * SEQ + j] = ebuf[j] * inv;
    }
}

// ---------- generic bf16 WMMA GEMM: out[M,N] = A[M,K] @ W[K,N] + bias (+epilogue) ----------
// WG tile 128x128 (8 waves as 4x2), wave tile 32x64 (2x4 WMMA tiles), K-chunk 32.
// MODE 0: bias only; MODE 1: bias + exact GELU; MODE 2: bias + residual add
template <int MODE>
__global__ __launch_bounds__(256) void gemm_bf16(const float* __restrict__ A,
                                                 const float* __restrict__ W,
                                                 const float* __restrict__ bias,
                                                 const float* __restrict__ res,
                                                 float* __restrict__ out,
                                                 int M, int N, int K) {
    __shared__ __align__(16) __bf16 a_lds[128 * 32];   // A tile   [128 m][32 k]
    __shared__ __align__(16) __bf16 wt_lds[128 * 32];  // W^T tile [128 n][32 k]
    int tid = threadIdx.x;
    int lane = tid & 31, wave = tid >> 5;
    int wm = wave >> 1, wn = wave & 1;  // 4x2 wave grid, 32x64 per wave
    int tileM = blockIdx.y * 128, tileN = blockIdx.x * 128;
    int half = lane >> 4, lm = lane & 15;

    v8f acc[2][4] = {};

    for (int k0 = 0; k0 < K; k0 += 32) {
        // stage A (128x32) as 512 octets, 2 per thread: b128 loads + b128 LDS store
#pragma unroll
        for (int g = 0; g < 2; ++g) {
            int o = g * 256 + tid;
            int m = o >> 2, kk = (o & 3) * 8;
            stage8(A + (size_t)(tileM + m) * K + k0 + kk, a_lds + m * 32 + kk);
        }
        // stage W^T (128n x 32k): 16 column-strided loads, 2 packed LDS b128 stores
        {
            int n = tid >> 1, kk = (tid & 1) * 16;
            const float* src = W + (size_t)(k0 + kk) * N + tileN + n;
            float t[16];
#pragma unroll
            for (int i = 0; i < 16; ++i) t[i] = src[(size_t)i * N];
            v8bf r0, r1;
#pragma unroll
            for (int i = 0; i < 8; ++i) { r0[i] = (__bf16)t[i]; r1[i] = (__bf16)t[i + 8]; }
            __bf16* dst = wt_lds + n * 32 + kk;
            *(v8bf*)dst = r0;
            *(v8bf*)(dst + 8) = r1;
        }
        __syncthreads();

        v16bf afr[2], bfr[4];
#pragma unroll
        for (int mi = 0; mi < 2; ++mi) {
            int row = wm * 32 + mi * 16 + lm;
            const __bf16* base = a_lds + row * 32 + (half ? 8 : 0);
            afr[mi] = ldfrag(base, base + 16);
        }
#pragma unroll
        for (int nj = 0; nj < 4; ++nj) {
            int n = wn * 64 + nj * 16 + lm;
            const __bf16* base = wt_lds + n * 32 + (half ? 8 : 0);
            bfr[nj] = ldfrag(base, base + 16);
        }
#pragma unroll
        for (int mi = 0; mi < 2; ++mi)
#pragma unroll
            for (int nj = 0; nj < 4; ++nj)
                acc[mi][nj] = __builtin_amdgcn_wmma_f32_16x16x32_bf16(
                    false, afr[mi], false, bfr[nj], (short)0, acc[mi][nj], false, false);
        __syncthreads();
    }

    // epilogue (stores coalesce across lanes: 16 consecutive cols per half-wave)
#pragma unroll
    for (int mi = 0; mi < 2; ++mi) {
#pragma unroll
        for (int nj = 0; nj < 4; ++nj) {
            int col = tileN + wn * 64 + nj * 16 + lm;
            float bv = bias[col];
#pragma unroll
            for (int r = 0; r < 8; ++r) {
                int row = tileM + wm * 32 + mi * 16 + r + half * 8;
                float v = acc[mi][nj][r] + bv;
                if constexpr (MODE == 1) v = 0.5f * v * (1.f + erff(v * 0.70710678f));
                if constexpr (MODE == 2) v += res[(size_t)row * N + col];
                out[(size_t)row * N + col] = v;
            }
        }
    }
}

// ---------- flash attention with gaussian bias + padding mask ----------
// grid: (SEQ/64, B*NH), block 128 (4 waves). wave w handles rows [rowBase+16w, +16)
__global__ __launch_bounds__(128) void attn_kernel(const float* __restrict__ q,
                                                   const float* __restrict__ k,
                                                   const float* __restrict__ v,
                                                   const int* __restrict__ mask,
                                                   const float* __restrict__ gbias,
                                                   float* __restrict__ out) {
    __shared__ __align__(16) __bf16 q_lds[64 * 64];
    __shared__ __align__(16) __bf16 k_lds[32 * 64];
    __shared__ __align__(16) __bf16 vt_lds[64 * 32];  // transposed: [d][t]
    __shared__ __align__(16) float p_lds[4][16 * 32];
    int tid = threadIdx.x, lane = tid & 31, wave = tid >> 5;
    int bh = blockIdx.y, b = bh / NH, h = bh % NH;
    int rowBase = blockIdx.x * 64;
    int hoff = h * DH;
    int half = lane >> 4, lm = lane & 15;

    // load q tile (64x64) as bf16: 512 octets, 4 per thread
#pragma unroll
    for (int g = 0; g < 4; ++g) {
        int o = g * 128 + tid;
        int m = o >> 3, d0 = (o & 7) * 8;
        stage8(q + ((size_t)(b * SEQ + rowBase + m)) * D_MODEL + hoff + d0,
               q_lds + m * 64 + d0);
    }

    float mreg[8], lreg[8];
    v8f oacc[4] = {};
#pragma unroll
    for (int r = 0; r < 8; ++r) { mreg[r] = -1e30f; lreg[r] = 0.f; }

    for (int c = 0; c < SEQ / 32; ++c) {
        int t0 = c * 32;
        // stage k (32x64) row-major and v (32x64) transposed into [d][t]
#pragma unroll
        for (int g = 0; g < 2; ++g) {
            int o = g * 128 + tid;
            int t = o >> 3, d0 = (o & 7) * 8;
            size_t gb = ((size_t)(b * SEQ + t0 + t)) * D_MODEL + hoff + d0;
            stage8(k + gb, k_lds + t * 64 + d0);
            float4 f0 = *(const float4*)(v + gb);
            float4 f1 = *(const float4*)(v + gb + 4);
            float tv[8] = {f0.x, f0.y, f0.z, f0.w, f1.x, f1.y, f1.z, f1.w};
#pragma unroll
            for (int i = 0; i < 8; ++i) vt_lds[(d0 + i) * 32 + t] = (__bf16)tv[i];
        }
        __syncthreads();

        // scores: q(16x64) @ k^T(64x32) -> two 16x16 tiles
        float s[2][8];
#pragma unroll
        for (int hf = 0; hf < 2; ++hf) {
            v8f sa = {};
#pragma unroll
            for (int st = 0; st < 2; ++st) {
                int qrow = wave * 16 + lm;
                const __bf16* ab = q_lds + qrow * 64 + st * 32 + (half ? 8 : 0);
                v16bf af = ldfrag(ab, ab + 16);
                int kr = hf * 16 + lm;
                const __bf16* bb = k_lds + kr * 64 + st * 32 + (half ? 8 : 0);
                v16bf bf = ldfrag(bb, bb + 16);
                sa = __builtin_amdgcn_wmma_f32_16x16x32_bf16(false, af, false, bf,
                                                             (short)0, sa, false, false);
            }
            int tcol = t0 + hf * 16 + lm;
            float mterm = -10000.f * (1.f - (float)mask[b * SEQ + tcol]);
#pragma unroll
            for (int r = 0; r < 8; ++r) {
                int irow = rowBase + wave * 16 + r + half * 8;
                s[hf][r] = sa[r] * 0.125f + gbias[(size_t)irow * SEQ + tcol] + mterm;
            }
        }

        // online softmax update
#pragma unroll
        for (int r = 0; r < 8; ++r) {
            float tm = redmax16(fmaxf(s[0][r], s[1][r]));
            float mn = fmaxf(mreg[r], tm);
            float al = __expf(mreg[r] - mn);
            float p0 = __expf(s[0][r] - mn);
            float p1 = __expf(s[1][r] - mn);
            float rs = redsum16(p0 + p1);
            lreg[r] = lreg[r] * al + rs;
            mreg[r] = mn;
            s[0][r] = p0; s[1][r] = p1;
#pragma unroll
            for (int j = 0; j < 4; ++j) oacc[j][r] *= al;
        }

        // C-layout -> row-major P in LDS (per-wave buffer)
#pragma unroll
        for (int hf = 0; hf < 2; ++hf)
#pragma unroll
            for (int r = 0; r < 8; ++r)
                p_lds[wave][(r + half * 8) * 32 + hf * 16 + lm] = s[hf][r];
        __syncthreads();

        // P(16x32) @ V(32x64): one A-frag, 4 WMMAs
        v16bf pf;
        {
            const float* pb = p_lds[wave] + lm * 32 + (half ? 8 : 0);
            float4 f0 = *(const float4*)pb;
            float4 f1 = *(const float4*)(pb + 4);
            float4 f2 = *(const float4*)(pb + 16);
            float4 f3 = *(const float4*)(pb + 20);
            pf[0]  = (__bf16)f0.x; pf[1]  = (__bf16)f0.y; pf[2]  = (__bf16)f0.z; pf[3]  = (__bf16)f0.w;
            pf[4]  = (__bf16)f1.x; pf[5]  = (__bf16)f1.y; pf[6]  = (__bf16)f1.z; pf[7]  = (__bf16)f1.w;
            pf[8]  = (__bf16)f2.x; pf[9]  = (__bf16)f2.y; pf[10] = (__bf16)f2.z; pf[11] = (__bf16)f2.w;
            pf[12] = (__bf16)f3.x; pf[13] = (__bf16)f3.y; pf[14] = (__bf16)f3.z; pf[15] = (__bf16)f3.w;
        }
#pragma unroll
        for (int j = 0; j < 4; ++j) {
            const __bf16* vb = vt_lds + (j * 16 + lm) * 32 + (half ? 8 : 0);
            v16bf vf = ldfrag(vb, vb + 16);
            oacc[j] = __builtin_amdgcn_wmma_f32_16x16x32_bf16(false, pf, false, vf,
                                                              (short)0, oacc[j], false, false);
        }
        __syncthreads();
    }

    // normalize and write out [B,S,D]
#pragma unroll
    for (int r = 0; r < 8; ++r) {
        float inv = 1.f / lreg[r];
        int irow = rowBase + wave * 16 + r + half * 8;
        float* op = out + ((size_t)(b * SEQ + irow)) * D_MODEL + hoff;
#pragma unroll
        for (int j = 0; j < 4; ++j) op[j * 16 + lm] = oacc[j][r] * inv;
    }
}

// ---------- layernorm over rows of 768 ----------
__global__ __launch_bounds__(256) void ln_kernel(const float* __restrict__ in,
                                                 const float* __restrict__ gamma,
                                                 const float* __restrict__ beta,
                                                 float* __restrict__ out) {
    __shared__ float red[256], red2[256];
    int row = blockIdx.x, tid = threadIdx.x;
    const float* x = in + (size_t)row * D_MODEL;
    float v[3], a = 0.f, bq = 0.f;
#pragma unroll
    for (int i = 0; i < 3; ++i) {
        v[i] = x[tid + i * 256];
        a += v[i];
        bq += v[i] * v[i];
    }
    red[tid] = a; red2[tid] = bq;
    __syncthreads();
    for (int s = 128; s > 0; s >>= 1) {
        if (tid < s) { red[tid] += red[tid + s]; red2[tid] += red2[tid + s]; }
        __syncthreads();
    }
    float u = red[0] * (1.f / 768.f);
    float var = red2[0] * (1.f / 768.f) - u * u;
    float inv = rsqrtf(var + 1e-12f);
#pragma unroll
    for (int i = 0; i < 3; ++i) {
        int c = tid + i * 256;
        out[(size_t)row * D_MODEL + c] = gamma[c] * (v[i] - u) * inv + beta[c];
    }
}

// ---------- orchestration ----------
extern "C" void kernel_launch(void* const* d_in, const int* in_sizes, int n_in,
                              void* d_out, int out_size, void* d_ws, size_t ws_size,
                              hipStream_t stream) {
    const float* x      = (const float*)d_in[0];
    const float* wq     = (const float*)d_in[1];
    const float* bq     = (const float*)d_in[2];
    const float* wk     = (const float*)d_in[3];
    const float* bk     = (const float*)d_in[4];
    const float* wv     = (const float*)d_in[5];
    const float* bv     = (const float*)d_in[6];
    const float* w_proj = (const float*)d_in[7];
    const float* b_proj = (const float*)d_in[8];
    const float* gamma1 = (const float*)d_in[9];
    const float* beta1  = (const float*)d_in[10];
    const float* w1     = (const float*)d_in[11];
    const float* b1     = (const float*)d_in[12];
    const float* w2     = (const float*)d_in[13];
    const float* b2     = (const float*)d_in[14];
    const float* gamma2 = (const float*)d_in[15];
    const float* beta2  = (const float*)d_in[16];
    const int*   mask   = (const int*)d_in[17];

    float* ws = (float*)d_ws;
    const size_t M   = 8 * SEQ;                 // 4096
    const size_t SZ  = M * D_MODEL;             // 3,145,728 floats
    const size_t OG  = 0;                       // gaussian bias [512,512]
    const size_t OQ  = 262144;
    const size_t OK  = OQ + SZ;
    const size_t OV  = OK + SZ;
    const size_t OA  = OV + SZ;                 // attention output
    const size_t OT  = OA + SZ;                 // tmp (pre-LN) buffer, reused
    const size_t OH1 = OT + SZ;                 // post-LN1 hidden
    const size_t OF1 = OQ;                      // FFN1 out (4096x3072) overlays dead q/k/v/a

    // 1. gaussian bias table
    gbias_kernel<<<SEQ, 256, 0, stream>>>(ws + OG);

    // 2. Q/K/V projections
    dim3 gD(D_MODEL / 128, M / 128);
    gemm_bf16<0><<<gD, 256, 0, stream>>>(x, wq, bq, nullptr, ws + OQ, M, D_MODEL, D_MODEL);
    gemm_bf16<0><<<gD, 256, 0, stream>>>(x, wk, bk, nullptr, ws + OK, M, D_MODEL, D_MODEL);
    gemm_bf16<0><<<gD, 256, 0, stream>>>(x, wv, bv, nullptr, ws + OV, M, D_MODEL, D_MODEL);

    // 3. attention
    attn_kernel<<<dim3(SEQ / 64, 8 * NH), 128, 0, stream>>>(ws + OQ, ws + OK, ws + OV,
                                                            mask, ws + OG, ws + OA);

    // 4. output projection + residual(x) -> tmp; LN1 -> h1
    gemm_bf16<2><<<gD, 256, 0, stream>>>(ws + OA, w_proj, b_proj, x, ws + OT, M, D_MODEL, D_MODEL);
    ln_kernel<<<M, 256, 0, stream>>>(ws + OT, gamma1, beta1, ws + OH1);

    // 5. FFN1 (768->3072, GELU) -> ff1; FFN2 (3072->768) + residual(h1) -> tmp; LN2 -> out
    gemm_bf16<1><<<dim3(3072 / 128, M / 128), 256, 0, stream>>>(ws + OH1, w1, b1, nullptr,
                                                                ws + OF1, M, 3072, D_MODEL);
    gemm_bf16<2><<<gD, 256, 0, stream>>>(ws + OF1, w2, b2, ws + OH1, ws + OT, M, D_MODEL, 3072);
    ln_kernel<<<M, 256, 0, stream>>>(ws + OT, gamma2, beta2, (float*)d_out);
}
